// StringGNNPerturbModel_6923487281766
// MI455X (gfx1250) — matrix operand order repaired and checked by reading, not compiled, not run on hardware
//
#include <hip/hip_runtime.h>
#include <hip/hip_bf16.h>

// ---------------------------------------------------------------------------
// CDNA5 (gfx1250) StringGNNPerturbModel inference.
//  - Dense GEMMs: v_wmma_f32_16x16x32_bf16, fragments staged in LDS in
//    fragment order so each lane loads its 32B with 2x ds_load_b128.
//  - Staging loads are branch-free (index clamping instead of exec guards).
//  - Edge scatter-sum: wave32 per edge, float4 gathers + global_atomic_add_f32.
// ---------------------------------------------------------------------------

typedef __attribute__((ext_vector_type(16))) __bf16    v16bf;
typedef __attribute__((ext_vector_type(8)))  float     v8f;
typedef __attribute__((ext_vector_type(8)))  unsigned  u32x8;

#define N_NODES 20000
#define N_EDGES 640000
#define BATCH   256
#define GNN_DIM 256
#define HID     512
#define INNER   2048
#define RANK    512
#define N_CLASSES 3
#define N_GENES 6640

// ---- pack two f32 into one u32 of bf16 (round-to-nearest-even) ----
__device__ __forceinline__ unsigned pack_bf2(float x, float y) {
    union { float f; unsigned u; } a, b; a.f = x; b.f = y;
    unsigned ra = a.u + 0x7fffu + ((a.u >> 16) & 1u);
    unsigned rb = b.u + 0x7fffu + ((b.u >> 16) & 1u);
    return (ra >> 16) | (rb & 0xffff0000u);
}

// ---------------------------------------------------------------------------
// LayerNorm: one wave32 per row.
// ---------------------------------------------------------------------------
__global__ void ln_rows(const float* __restrict__ x,
                        const float* __restrict__ s,
                        const float* __restrict__ b,
                        float* __restrict__ y, int rows, int D) {
    int wid  = (blockIdx.x * blockDim.x + threadIdx.x) >> 5;
    int lane = threadIdx.x & 31;
    if (wid >= rows) return;
    const float* xr = x + (long)wid * D;
    float sum = 0.f, sq = 0.f;
    for (int i = lane; i < D; i += 32) { float v = xr[i]; sum += v; sq += v * v; }
    #pragma unroll
    for (int o = 16; o; o >>= 1) {
        sum += __shfl_xor(sum, o, 32);
        sq  += __shfl_xor(sq,  o, 32);
    }
    float mu  = sum / (float)D;
    float var = sq / (float)D - mu * mu;
    float inv = rsqrtf(var + 1e-5f);
    float* yr = y + (long)wid * D;
    for (int i = lane; i < D; i += 32)
        yr[i] = (xr[i] - mu) * inv * s[i] + b[i];
}

// ---------------------------------------------------------------------------
// Weighted edge scatter-sum: agg[dst] += hln[src] * w. One wave32 per edge,
// each lane moves 2x float4 (256 dims). Atomics land in L2 (agg ~20.5 MB).
// ---------------------------------------------------------------------------
__global__ void scatter_edges(const int* __restrict__ ei,
                              const float* __restrict__ ew,
                              const float* __restrict__ hln,
                              float* __restrict__ agg, int nE) {
    int e    = blockIdx.x * (blockDim.x >> 5) + (threadIdx.x >> 5);
    int lane = threadIdx.x & 31;
    if (e >= nE) return;
    int   src = ei[e];
    int   dst = ei[nE + e];
    float w   = ew[e];
    const float4* hv = (const float4*)(hln + (long)src * GNN_DIM);
    float* p = agg + (long)dst * GNN_DIM;
    #pragma unroll
    for (int i = 0; i < 2; ++i) {
        int q = lane + i * 32;          // which float4 of 64
        float4 v = hv[q];
        int o = q * 4;
        atomicAdd(p + o + 0, v.x * w);
        atomicAdd(p + o + 1, v.y * w);
        atomicAdd(p + o + 2, v.z * w);
        atomicAdd(p + o + 3, v.w * w);
    }
}

// ---------------------------------------------------------------------------
// OOV-safe perturbation gather.
// ---------------------------------------------------------------------------
__global__ void gather_pert(const int* __restrict__ idx,
                            const float* __restrict__ xpost,
                            const float* __restrict__ oov,
                            float* __restrict__ pert) {
    int b = blockIdx.x, t = threadIdx.x;
    int id = idx[b];
    pert[(long)b * GNN_DIM + t] = (id < 0) ? oov[t] : xpost[(long)id * GNN_DIM + t];
}

// ---------------------------------------------------------------------------
// bf16-WMMA GEMM: C = act(A[M,K] @ W + bias) (+ residual)
//   BT=false: W row-major [K,N];  BT=true: W row-major [N,K] (B transposed).
// Block: 256 threads = 8 waves (2 M-groups x 4 N-groups), each wave owns a
// 32x32 output (2x2 of 16x16 WMMA tiles) -> 4 v_wmma per wave per K-step.
// Block tile: 64(M) x 128(N) x 32(K). K must be a multiple of 32.
// LDS holds tiles pre-swizzled into fragment order: per (tile, lane) the 16
// bf16 fragment values are 32 contiguous bytes -> 2x ds_load_b128 per frag.
// Out-of-range staging indices are CLAMPED (branch-free loads); the garbage
// they produce only lands in output rows/cols that the epilogue never stores.
// ACT: 0=none, 1=relu, 2=gelu(erf)
// ---------------------------------------------------------------------------
#define WMMA_BF16(A_, B_, C_)                                              \
    __builtin_amdgcn_wmma_f32_16x16x32_bf16(false, (A_), false, (B_),      \
                                            (short)0, (C_), false, false)

template <int ACT, bool BT>
__global__ __launch_bounds__(256)
void gemm_bf16_wmma(const float* __restrict__ A,
                    const float* __restrict__ W,
                    const float* __restrict__ bias,
                    const float* __restrict__ residual,
                    float* __restrict__ C,
                    int M, int N, int K) {
    // fragment-order LDS: [tile][lane][8 x u32]
    __shared__ __align__(32) unsigned lds_a[4 * 32 * 8];   // 4 M-tiles  (4 KB)
    __shared__ __align__(32) unsigned lds_b[8 * 32 * 8];   // 8 N-tiles  (8 KB)

    const int tid  = threadIdx.x;
    const int lane = tid & 31;
    const int wid  = tid >> 5;
    const int wm   = wid & 1;            // 0..1 : 32-row group
    const int wn   = wid >> 1;           // 0..3 : 32-col group
    const int blockM = blockIdx.y * 64;
    const int blockN = blockIdx.x * 128;

    v8f acc00 = {}, acc01 = {}, acc10 = {}, acc11 = {};

    for (int kt = 0; kt < K; kt += 32) {
        // ---- stage A tile (64 rows x 32 k) as bf16 pairs in fragment order
        // A-frag layout: lane = m + 16*((k>>3)&1); elem = 8*(k>>4) + (k&7)
        #pragma unroll
        for (int i = 0; i < 4; ++i) {
            int e  = i * 256 + tid;       // 0..1023 u32 slots
            int r  = e >> 4;              // row 0..63
            int k0 = (e & 15) * 2;        // even k
            int gr = blockM + r;  if (gr >= M) gr = M - 1;   // clamp (branch-free)
            float2 v = *(const float2*)(A + (long)gr * K + kt + k0);
            int mt   = r >> 4;
            int lt   = (r & 15) + (((k0 >> 3) & 1) << 4);
            int elem = ((k0 >> 4) << 3) + (k0 & 7);
            lds_a[(mt * 32 + lt) * 8 + (elem >> 1)] = pack_bf2(v.x, v.y);
        }
        // ---- stage B tile (32 k x 128 n) in fragment order
        // B-frag layout: lane = n + 16*((k>>3)&1); elem = 8*(k>>4) + (k&7)
        #pragma unroll
        for (int i = 0; i < 8; ++i) {
            int e  = i * 256 + tid;       // 0..2047 u32 slots
            int n  = e & 127;
            int k0 = (e >> 7) * 2;        // even k
            int gn = blockN + n;  if (gn >= N) gn = N - 1;   // clamp (branch-free)
            int gk = kt + k0;
            float vx, vy;
            if (BT) {
                float2 v = *(const float2*)(W + (long)gn * K + gk);
                vx = v.x; vy = v.y;
            } else {
                vx = W[(long)gk * N + gn];
                vy = W[(long)(gk + 1) * N + gn];
            }
            int nt   = n >> 4;
            int lt   = (n & 15) + (((k0 >> 3) & 1) << 4);
            int elem = ((k0 >> 4) << 3) + (k0 & 7);
            lds_b[(nt * 32 + lt) * 8 + (elem >> 1)] = pack_bf2(vx, vy);
        }
        __syncthreads();

        // ---- fragment loads: 32 contiguous bytes per lane (2x ds_load_b128)
        const u32x8* pa = (const u32x8*)lds_a;
        const u32x8* pb = (const u32x8*)lds_b;
        v16bf a0 = __builtin_bit_cast(v16bf, pa[(wm * 2 + 0) * 32 + lane]);
        v16bf a1 = __builtin_bit_cast(v16bf, pa[(wm * 2 + 1) * 32 + lane]);
        v16bf b0 = __builtin_bit_cast(v16bf, pb[(wn * 2 + 0) * 32 + lane]);
        v16bf b1 = __builtin_bit_cast(v16bf, pb[(wn * 2 + 1) * 32 + lane]);

        acc00 = WMMA_BF16(a0, b0, acc00);
        acc01 = WMMA_BF16(a0, b1, acc01);
        acc10 = WMMA_BF16(a1, b0, acc10);
        acc11 = WMMA_BF16(a1, b1, acc11);

        __syncthreads();
    }

    // ---- epilogue: bias + activation + residual; D layout: VGPR r ->
    // m = r + 8*(lane>>4), n = lane&15 within each 16x16 tile.
    const int rbase = blockM + wm * 32 + ((lane >> 4) ? 8 : 0);
    const int cbase = blockN + wn * 32 + (lane & 15);
    #pragma unroll
    for (int ti = 0; ti < 2; ++ti) {
        #pragma unroll
        for (int tj = 0; tj < 2; ++tj) {
            const v8f& acc = (ti == 0) ? (tj == 0 ? acc00 : acc01)
                                       : (tj == 0 ? acc10 : acc11);
            int col = cbase + tj * 16;
            if (col >= N) continue;
            float bv = bias ? bias[col] : 0.f;
            #pragma unroll
            for (int r = 0; r < 8; ++r) {
                int row = rbase + ti * 16 + r;
                if (row < M) {
                    float v = acc[r] + bv;
                    if (ACT == 1)      v = fmaxf(v, 0.f);
                    else if (ACT == 2) v = 0.5f * v * (1.f + erff(v * 0.70710678118f));
                    if (residual) v += residual[(long)row * N + col];
                    C[(long)row * N + col] = v;
                }
            }
        }
    }
}

// ---------------------------------------------------------------------------
// Host-side orchestration.
// ---------------------------------------------------------------------------
extern "C" void kernel_launch(void* const* d_in, const int* in_sizes, int n_in,
                              void* d_out, int out_size, void* d_ws, size_t ws_size,
                              hipStream_t stream) {
    (void)in_sizes; (void)n_in; (void)out_size; (void)ws_size;

    const int*   node_indices = (const int*)  d_in[0];
    const int*   edge_index   = (const int*)  d_in[1];
    const float* edge_weight  = (const float*)d_in[2];
    const float* partial_emb  = (const float*)d_in[3];
    const float* oov_emb      = (const float*)d_in[4];
    const float* gnn_ln_s     = (const float*)d_in[5];
    const float* gnn_ln_b     = (const float*)d_in[6];
    const float* gnn_w        = (const float*)d_in[7];
    const float* gnn_b        = (const float*)d_in[8];
    const float* post_w       = (const float*)d_in[9];
    const float* post_b       = (const float*)d_in[10];
    const float* pin_w        = (const float*)d_in[11];
    const float* pin_b        = (const float*)d_in[12];
    const float* blk_ln_s     = (const float*)d_in[13];
    const float* blk_ln_b     = (const float*)d_in[14];
    const float* blk_w1       = (const float*)d_in[15];
    const float* blk_b1       = (const float*)d_in[16];
    const float* blk_w2       = (const float*)d_in[17];
    const float* blk_b2       = (const float*)d_in[18];
    const float* pout_w       = (const float*)d_in[19];
    const float* pout_b       = (const float*)d_in[20];
    const float* gene_emb     = (const float*)d_in[21];
    float* out = (float*)d_out;

    // workspace layout (floats)
    float* ws    = (float*)d_ws;
    float* xbuf  = ws;                                    // 20000*256
    float* hln   = xbuf  + (long)N_NODES * GNN_DIM;       // 20000*256
    float* agg   = hln   + (long)N_NODES * GNN_DIM;       // 20000*256
    float* xpost = agg   + (long)N_NODES * GNN_DIM;       // 20000*256
    float* pert  = xpost + (long)N_NODES * GNN_DIM;       // 256*256
    float* hbuf  = pert  + (long)BATCH * GNN_DIM;         // 256*512
    float* zln   = hbuf  + (long)BATCH * HID;             // 256*512
    float* zbuf  = zln   + (long)BATCH * HID;             // 256*2048
    float* proj  = zbuf  + (long)BATCH * INNER;           // 256*1536

    hipMemcpyAsync(xbuf, partial_emb, (size_t)N_NODES * GNN_DIM * sizeof(float),
                   hipMemcpyDeviceToDevice, stream);

    dim3 blk256(256);
    // ---- GNN tail: 3 x (LN -> scatter-sum -> GEMM+relu + residual) ----
    for (int i = 0; i < 3; ++i) {
        ln_rows<<<(N_NODES + 7) / 8, blk256, 0, stream>>>(
            xbuf, gnn_ln_s + i * GNN_DIM, gnn_ln_b + i * GNN_DIM, hln, N_NODES, GNN_DIM);

        hipMemsetAsync(agg, 0, (size_t)N_NODES * GNN_DIM * sizeof(float), stream);

        scatter_edges<<<(N_EDGES + 7) / 8, blk256, 0, stream>>>(
            edge_index, edge_weight, hln, agg, N_EDGES);

        dim3 grid((GNN_DIM + 127) / 128, (N_NODES + 63) / 64);
        gemm_bf16_wmma<1, false><<<grid, blk256, 0, stream>>>(
            agg, gnn_w + (long)i * GNN_DIM * GNN_DIM, gnn_b + i * GNN_DIM,
            xbuf, xbuf, N_NODES, GNN_DIM, GNN_DIM);
    }

    // ---- post_mp ----
    {
        dim3 grid((GNN_DIM + 127) / 128, (N_NODES + 63) / 64);
        gemm_bf16_wmma<0, false><<<grid, blk256, 0, stream>>>(
            xbuf, post_w, post_b, nullptr, xpost, N_NODES, GNN_DIM, GNN_DIM);
    }

    // ---- OOV-safe gather ----
    gather_pert<<<BATCH, GNN_DIM, 0, stream>>>(node_indices, xpost, oov_emb, pert);

    // ---- head in-proj ----
    {
        dim3 grid((HID + 127) / 128, (BATCH + 63) / 64);
        gemm_bf16_wmma<0, false><<<grid, blk256, 0, stream>>>(
            pert, pin_w, pin_b, nullptr, hbuf, BATCH, HID, GNN_DIM);
    }

    // ---- 6 residual MLP blocks ----
    for (int i = 0; i < 6; ++i) {
        ln_rows<<<(BATCH + 7) / 8, blk256, 0, stream>>>(
            hbuf, blk_ln_s + i * HID, blk_ln_b + i * HID, zln, BATCH, HID);

        dim3 g1((INNER + 127) / 128, (BATCH + 63) / 64);
        gemm_bf16_wmma<2, false><<<g1, blk256, 0, stream>>>(
            zln, blk_w1 + (long)i * HID * INNER, blk_b1 + i * INNER,
            nullptr, zbuf, BATCH, INNER, HID);

        dim3 g2((HID + 127) / 128, (BATCH + 63) / 64);
        gemm_bf16_wmma<0, false><<<g2, blk256, 0, stream>>>(
            zbuf, blk_w2 + (long)i * INNER * HID, blk_b2 + i * HID,
            hbuf, hbuf, BATCH, HID, INNER);
    }

    // ---- out-proj: [256,512] @ [512,1536] ----
    {
        dim3 grid((N_CLASSES * RANK + 127) / 128, (BATCH + 63) / 64);
        gemm_bf16_wmma<0, false><<<grid, blk256, 0, stream>>>(
            hbuf, pout_w, pout_b, nullptr, proj, BATCH, N_CLASSES * RANK, HID);
    }

    // ---- logits: [768,512] @ [6640,512]^T ----
    {
        dim3 grid((N_GENES + 127) / 128, (BATCH * N_CLASSES + 63) / 64);
        gemm_bf16_wmma<0, true><<<grid, blk256, 0, stream>>>(
            proj, gene_emb, nullptr, nullptr, out,
            BATCH * N_CLASSES, N_GENES, RANK);
    }
}
